// DPINet_65618510348331
// MI455X (gfx1250) — matrix operand order repaired and checked by reading, not compiled, not run on hardware
//
#include <hip/hip_runtime.h>
#include <math.h>

// ---------------- constants from the reference ----------------
#define NPART   50000
#define NRELS   300000
#define BRIGID  5000
#define NFLUID  (NPART - BRIGID)
#define PSTEPS  2
#define NF_E    200

// padded row strides (multiples of 16 floats)
#define SP  16    // pe_in (13 -> 16)
#define SR  32    // rel_in (27 -> 32)
#define SE  208   // 200-wide activations
#define SH  304   // 300-wide activations

typedef _Float16 vhalf16 __attribute__((ext_vector_type(16)));
typedef float    vfloat8 __attribute__((ext_vector_type(8)));

// =====================================================================
// Weight packing: f32 [Kw x Nout] -> zero-padded f16 [Ka x Nstride],
// with up to 3 virtual row segments (for the fused concat GEMMs).
// =====================================================================
struct PackP {
  const float* src;
  _Float16* dst;
  int Nout, Nstride, Ka;
  int v0, l0, r0;
  int v1, l1, r1;
  int v2, l2, r2;
};

__global__ void k_pack_w(PackP p) {
  int i = blockIdx.x * blockDim.x + threadIdx.x;
  long total = (long)p.Ka * p.Nstride;
  if (i >= total) return;
  int v = i / p.Nstride, n = i % p.Nstride;
  int srcrow = -1;
  if (v >= p.v0 && v < p.v0 + p.l0) srcrow = p.r0 + (v - p.v0);
  else if (v >= p.v1 && v < p.v1 + p.l1) srcrow = p.r1 + (v - p.v1);
  else if (v >= p.v2 && v < p.v2 + p.l2) srcrow = p.r2 + (v - p.v2);
  float val = 0.f;
  if (srcrow >= 0 && n < p.Nout) val = p.src[(long)srcrow * p.Nout + n];
  p.dst[i] = (_Float16)val;
}

// =====================================================================
// WMMA GEMM: out[M x Nstride] = act( concatA[M x Ka] @ Wh[Ka x Nstride] + bias )
// A is a virtual concat of up to 3 padded-f32 sources with optional row gather.
// Segment boundaries (vend) must be multiples of 16.
// =====================================================================
struct Seg { const float* a; const int* idx; int vend; int stride; };

struct GemmP {
  Seg s0, s1, s2;
  const _Float16* Wh;   // packed [Ka x Nstride], pads zeroed
  const float* bias;    // [Nout]
  float* out;           // [M x Nstride] (pad cols written as 0)
  const int* outIdx;    // optional row scatter
  int M, Nout, Nstride, Ka, doRelu;
};

#define TBM 128
#define TBN 128
#define TBK 32
#define LDK 34

union U16x8 { uint4 u; _Float16 h[8]; };

__launch_bounds__(256, 2)
__global__ void wmma_gemm_k(GemmP p) {
  __shared__ _Float16 As[TBM][LDK];   // [m][k]
  __shared__ _Float16 Bs[TBN][LDK];   // [n][k] transposed
  const int tid  = threadIdx.x;
  const int lane = tid & 31;
  const int wave = tid >> 5;
  const int wr   = wave >> 1;          // 0..3: rows 32*wr
  const int wcb  = (wave & 1) * 64;    // col base 0/64
  const int bm   = blockIdx.y * TBM;
  const int bn   = blockIdx.x * TBN;
  const int hi   = lane >> 4;
  const int l15  = lane & 15;

  // A-tile job: one 16-float aligned run per thread
  const int am = tid >> 1;
  const int ac = (tid & 1) * 16;
  // B-tile job: one weight row x 16 halves per thread
  const int bk  = tid >> 3;
  const int bnc = (tid & 7) * 16;

  vfloat8 acc[2][4] = {};

  for (int k0 = 0; k0 < p.Ka; k0 += TBK) {
    // ---- A tile load: 4x float4, branchless row guard, uniform segment pick
    {
      int c16 = k0 + ac;
      int r = bm + am;
      if (c16 < p.Ka) {
        const float* base; const int* gidx; int stride, off;
        if (c16 < p.s0.vend)      { base = p.s0.a; gidx = p.s0.idx; stride = p.s0.stride; off = c16; }
        else if (c16 < p.s1.vend) { base = p.s1.a; gidx = p.s1.idx; stride = p.s1.stride; off = c16 - p.s0.vend; }
        else                      { base = p.s2.a; gidx = p.s2.idx; stride = p.s2.stride; off = c16 - p.s1.vend; }
        int rc = r < p.M ? r : p.M - 1;
        int rr = gidx ? gidx[rc] : rc;
        const float4* ap = (const float4*)(base + (long)rr * stride + off);
        bool ok = (r < p.M);
#pragma unroll
        for (int j = 0; j < 4; ++j) {
          float4 v = ap[j];
          As[am][ac + 4 * j + 0] = (_Float16)(ok ? v.x : 0.f);
          As[am][ac + 4 * j + 1] = (_Float16)(ok ? v.y : 0.f);
          As[am][ac + 4 * j + 2] = (_Float16)(ok ? v.z : 0.f);
          As[am][ac + 4 * j + 3] = (_Float16)(ok ? v.w : 0.f);
        }
      } else {
#pragma unroll
        for (int j = 0; j < 16; ++j) As[am][ac + j] = (_Float16)0.f;
      }
    }
    // ---- B tile load: 2x uint4 of packed f16 weights, transposed store
    {
      int k = k0 + bk;
      int gn = bn + bnc;
      if (k < p.Ka && gn < p.Nstride) {
        const uint4* wp = (const uint4*)(p.Wh + (long)k * p.Nstride + gn);
        U16x8 a, b;
        a.u = wp[0]; b.u = wp[1];
#pragma unroll
        for (int j = 0; j < 8; ++j) {
          Bs[bnc + j][bk]     = a.h[j];
          Bs[bnc + 8 + j][bk] = b.h[j];
        }
      } else {
#pragma unroll
        for (int j = 0; j < 16; ++j) Bs[bnc + j][bk] = (_Float16)0.f;
      }
    }
    __syncthreads();

    // ---- fragments per ISA 7.12.2 (wave32), paired 32-bit LDS reads
    vhalf16 afr[2], bfr[4];
#pragma unroll
    for (int mt = 0; mt < 2; ++mt) {
      int m = wr * 32 + mt * 16 + l15;
      union { vhalf16 v; unsigned u[8]; } ua;
#pragma unroll
      for (int j = 0; j < 8; ++j) {
        int kk = (hi << 3) + ((j < 4) ? (2 * j) : (2 * j + 8));
        ua.u[j] = *(const unsigned*)&As[m][kk];
      }
      afr[mt] = ua.v;
    }
#pragma unroll
    for (int nt = 0; nt < 4; ++nt) {
      int n = wcb + nt * 16 + l15;
      union { vhalf16 v; unsigned u[8]; } ub;
#pragma unroll
      for (int j = 0; j < 8; ++j) {
        int kk = (hi << 3) + ((j < 4) ? (2 * j) : (2 * j + 8));
        ub.u[j] = *(const unsigned*)&Bs[n][kk];
      }
      bfr[nt] = ub.v;
    }
#pragma unroll
    for (int mt = 0; mt < 2; ++mt)
#pragma unroll
      for (int nt = 0; nt < 4; ++nt)
        acc[mt][nt] = __builtin_amdgcn_wmma_f32_16x16x32_f16(
            false, afr[mt], false, bfr[nt], (short)0, acc[mt][nt], false, false);
    __syncthreads();
  }

  // ---- epilogue: bias + relu, zero pad columns, optional row scatter
#pragma unroll
  for (int mt = 0; mt < 2; ++mt) {
#pragma unroll
    for (int nt = 0; nt < 4; ++nt) {
      int col = bn + wcb + nt * 16 + l15;
      if (col >= p.Nstride) continue;
#pragma unroll
      for (int e = 0; e < 8; ++e) {
        int row = bm + wr * 32 + mt * 16 + (hi << 3) + e;
        if (row >= p.M) continue;
        float v = 0.f;
        if (col < p.Nout) {
          v = acc[mt][nt][e] + p.bias[col];
          if (p.doRelu && v < 0.f) v = 0.f;
        }
        int orow = p.outIdx ? p.outIdx[row] : row;
        p.out[(long)orow * p.Nstride + col] = v;
      }
    }
  }
}

// ---------------- auxiliary kernels ----------------
__global__ void k_mean_state(const float* __restrict__ state, float* mean6) {
  __shared__ float red[256][6];
  int t = threadIdx.x;
  float s[6] = {0, 0, 0, 0, 0, 0};
  for (int i = t; i < BRIGID; i += 256)
    for (int d = 0; d < 6; ++d) s[d] += state[i * 6 + d];
  for (int d = 0; d < 6; ++d) red[t][d] = s[d];
  __syncthreads();
  if (t < 6) {
    float tot = 0.f;
    for (int i = 0; i < 256; ++i) tot += red[i][t];
    mean6[t] = tot / (float)BRIGID;
  }
}

__global__ void k_attr_full(const float* attr, const float* state,
                            const float* mean6, float* attr_full) {
  int i = blockIdx.x * blockDim.x + threadIdx.x;
  if (i >= NPART) return;
  attr_full[i * 7] = attr[i];
  for (int d = 0; d < 6; ++d)
    attr_full[i * 7 + 1 + d] = (i < BRIGID) ? (state[i * 6 + d] - mean6[d]) : 0.f;
}

__global__ void k_pe_in(const float* attr_full, const float* state,
                        const int* nr, float* pe_in) {
  int i = blockIdx.x * blockDim.x + threadIdx.x;
  if (i >= NPART) return;
  int r = nr[i];
  float* o = pe_in + (long)i * SP;
  for (int d = 0; d < 7; ++d) o[d] = attr_full[r * 7 + d];
  for (int d = 0; d < 6; ++d) o[7 + d] = state[r * 6 + d];
  for (int d = 13; d < SP; ++d) o[d] = 0.f;
}

__global__ void k_relidx(const int* nr, const int* ns, const int* recv,
                         const int* send, int* ridx, int* sidx) {
  int i = blockIdx.x * blockDim.x + threadIdx.x;
  if (i >= NRELS) return;
  ridx[i] = nr[recv[i]];
  sidx[i] = ns[send[i]];
}

__global__ void k_rel_in(const float* attr_full, const float* state,
                         const float* Ra, const int* ridx, const int* sidx,
                         float* rel_in) {
  int i = blockIdx.x * blockDim.x + threadIdx.x;
  if (i >= NRELS) return;
  int rr = ridx[i], ss = sidx[i];
  float* o = rel_in + (long)i * SR;
  for (int d = 0; d < 7; ++d) o[d] = attr_full[rr * 7 + d];
  for (int d = 0; d < 7; ++d) o[7 + d] = attr_full[ss * 7 + d];
  for (int d = 0; d < 6; ++d) o[14 + d] = state[rr * 6 + d];
  for (int d = 0; d < 6; ++d) o[20 + d] = state[ss * 6 + d];
  o[26] = Ra[i];
  for (int d = 27; d < SR; ++d) o[d] = 0.f;
}

__global__ void k_zero(float* p, long n) {
  long stride = (long)gridDim.x * blockDim.x;
  for (long i = (long)blockIdx.x * blockDim.x + threadIdx.x; i < n; i += stride)
    p[i] = 0.f;
}

__global__ void k_scatter_add(const float* __restrict__ er,
                              const int* __restrict__ recv,
                              float* __restrict__ agg, long total) {
  long i = (long)blockIdx.x * blockDim.x + threadIdx.x;
  if (i >= total) return;
  int r = (int)(i / SE), c = (int)(i % SE);
  float v = er[i];
  if (v != 0.f) atomicAdd(&agg[(long)recv[r] * SE + c], v);
}

__global__ void k_rigid_head(const float* pe, const float* w0, const float* b0,
                             const float* w1, const float* b1, const float* w2,
                             const float* b2, const float* mean6,
                             const float* std_p, const float* mean_p,
                             float* small) {
  __shared__ float t0[NF_E], t1[NF_E], t2[NF_E], tt[7];
  int t = threadIdx.x;
  if (t < NF_E) {
    float s = 0.f;
    for (int i = 0; i < BRIGID; ++i) s += pe[(long)i * SE + t];
    t0[t] = s / (float)BRIGID;
  }
  __syncthreads();
  if (t < NF_E) {
    float s = b0[t];
    for (int k = 0; k < NF_E; ++k) s += t0[k] * w0[k * NF_E + t];
    t1[t] = s > 0.f ? s : 0.f;
  }
  __syncthreads();
  if (t < NF_E) {
    float s = b1[t];
    for (int k = 0; k < NF_E; ++k) s += t1[k] * w1[k * NF_E + t];
    t2[t] = s > 0.f ? s : 0.f;
  }
  __syncthreads();
  if (t < 7) {
    float s = b2[t];
    for (int k = 0; k < NF_E; ++k) s += t2[k] * w2[k * 7 + t];
    tt[t] = s;
  }
  __syncthreads();
  if (t == 0) {
    float w = tt[0], x = tt[1], y = tt[2], z = tt[3];
    float n = sqrtf(w * w + x * x + y * y + z * z);
    w /= n; x /= n; y /= n; z /= n;
    float R[9] = {1.f - 2.f * (y * y + z * z), 2.f * (x * y + z * w), 2.f * (x * z - y * w),
                  2.f * (x * y - z * w), 1.f - 2.f * (x * x + z * z), 2.f * (y * z + x * w),
                  2.f * (x * z + y * w), 2.f * (y * z - x * w), 1.f - 2.f * (x * x + y * y)};
    for (int i = 0; i < 9; ++i) small[i] = R[i];
    for (int d = 0; d < 3; ++d) small[9 + d] = tt[4 + d] * std_p[d];
    for (int d = 0; d < 3; ++d) small[12 + d] = mean6[d] * std_p[d] + mean_p[d];
  }
}

__global__ void k_rigid_out(const float* state, const float* small,
                            const float* std_p, const float* mean_p,
                            const float* mean_v, const float* std_v, float* out) {
  int i = blockIdx.x * blockDim.x + threadIdx.x;
  if (i >= BRIGID) return;
  float p0[3], pc[3];
  for (int d = 0; d < 3; ++d) {
    p0[d] = state[i * 6 + d] * std_p[d] + mean_p[d];
    pc[d] = p0[d] - small[12 + d];
  }
  for (int c = 0; c < 3; ++c) {
    float p1 = small[9 + c] + small[12 + c];
    for (int k = 0; k < 3; ++k) p1 += pc[k] * small[k * 3 + c];
    float v = (p1 - p0[c]) * 60.f;  // / DT
    out[i * 3 + c] = (v - mean_v[c]) / std_v[c];
  }
}

__global__ void k_fluid_out(const float* F2, const float* w, const float* b,
                            float* out) {
  int i = blockIdx.x * blockDim.x + threadIdx.x;
  if (i >= NFLUID * 3) return;
  int r = i / 3, c = i % 3;
  float s = b[c];
  const float* row = F2 + (long)r * SE;
  for (int k = 0; k < NF_E; ++k) s += row[k] * w[k * 3 + c];
  out[(long)(BRIGID + r) * 3 + c] = s;
}

// ---------------- host side ----------------
static void pack_w(hipStream_t s, const float* src, _Float16* dst, int Nout,
                   int Nstride, int Ka, int v0, int l0, int r0, int v1, int l1,
                   int r1, int v2, int l2, int r2) {
  PackP p{src, dst, Nout, Nstride, Ka, v0, l0, r0, v1, l1, r1, v2, l2, r2};
  long total = (long)Ka * Nstride;
  k_pack_w<<<(unsigned)((total + 255) / 256), 256, 0, s>>>(p);
}

static void gemm(hipStream_t s, Seg s0, Seg s1, Seg s2, const _Float16* Wh,
                 const float* bias, float* out, const int* outIdx, int M,
                 int Nout, int Nstride, int Ka, int doRelu) {
  GemmP p{s0, s1, s2, Wh, bias, out, outIdx, M, Nout, Nstride, Ka, doRelu};
  dim3 grid((Nstride + TBN - 1) / TBN, (M + TBM - 1) / TBM);
  wmma_gemm_k<<<grid, dim3(256), 0, s>>>(p);
}

extern "C" void kernel_launch(void* const* d_in, const int* in_sizes, int n_in,
                              void* d_out, int out_size, void* d_ws,
                              size_t ws_size, hipStream_t stream) {
  (void)in_sizes; (void)n_in; (void)out_size; (void)ws_size;
  const float* attr  = (const float*)d_in[0];
  const float* state = (const float*)d_in[1];
  const float* Ra    = (const float*)d_in[2];
  const int* rel_recv = (const int*)d_in[3];
  const int* rel_send = (const int*)d_in[4];
  const int* node_r   = (const int*)d_in[5];
  const int* node_s   = (const int*)d_in[6];
  const float* pe_w0 = (const float*)d_in[9];
  const float* pe_b0 = (const float*)d_in[10];
  const float* pe_w1 = (const float*)d_in[11];
  const float* pe_b1 = (const float*)d_in[12];
  const float* re_w0 = (const float*)d_in[13];
  const float* re_b0 = (const float*)d_in[14];
  const float* re_w1 = (const float*)d_in[15];
  const float* re_b1 = (const float*)d_in[16];
  const float* re_w2 = (const float*)d_in[17];
  const float* re_b2 = (const float*)d_in[18];
  const float* rp_w  = (const float*)d_in[19];
  const float* rp_b  = (const float*)d_in[20];
  const float* pp_w  = (const float*)d_in[21];
  const float* pp_b  = (const float*)d_in[22];
  const float* rg_w0 = (const float*)d_in[23];
  const float* rg_b0 = (const float*)d_in[24];
  const float* rg_w1 = (const float*)d_in[25];
  const float* rg_b1 = (const float*)d_in[26];
  const float* rg_w2 = (const float*)d_in[27];
  const float* rg_b2 = (const float*)d_in[28];
  const float* fl_w0 = (const float*)d_in[29];
  const float* fl_b0 = (const float*)d_in[30];
  const float* fl_w1 = (const float*)d_in[31];
  const float* fl_b1 = (const float*)d_in[32];
  const float* fl_w2 = (const float*)d_in[33];
  const float* fl_b2 = (const float*)d_in[34];
  const float* mean_v = (const float*)d_in[35];
  const float* std_v  = (const float*)d_in[36];
  const float* mean_p = (const float*)d_in[37];
  const float* std_p  = (const float*)d_in[38];
  float* out = (float*)d_out;

  // ---- workspace carve-up (256B aligned)
  char* wptr = (char*)d_ws;
  auto carve = [&](size_t bytes) -> void* {
    void* p = (void*)wptr;
    wptr += (bytes + 255) & ~(size_t)255;
    return p;
  };
  float* mean6     = (float*)carve(8 * sizeof(float));
  float* smallb    = (float*)carve(16 * sizeof(float));
  float* attr_full = (float*)carve((size_t)NPART * 7 * sizeof(float));
  float* pe_in     = (float*)carve((size_t)NPART * SP * sizeof(float));
  float* Ptmp      = (float*)carve((size_t)NPART * SE * sizeof(float));
  float* part_enc  = (float*)carve((size_t)NPART * SE * sizeof(float));
  float* part_eff  = (float*)carve((size_t)NPART * SE * sizeof(float));
  float* agg       = (float*)carve((size_t)NPART * SE * sizeof(float));
  int*   ridx      = (int*)carve((size_t)NRELS * sizeof(int));
  int*   sidx      = (int*)carve((size_t)NRELS * sizeof(int));
  _Float16* whPe0  = (_Float16*)carve((size_t)16 * SE * 2);
  _Float16* whPe1  = (_Float16*)carve((size_t)SE * SE * 2);
  _Float16* whRe0  = (_Float16*)carve((size_t)32 * SH * 2);
  _Float16* whRe1  = (_Float16*)carve((size_t)SH * SH * 2);
  _Float16* whRe2  = (_Float16*)carve((size_t)SH * SH * 2);
  _Float16* whRp   = (_Float16*)carve((size_t)720 * SE * 2);
  _Float16* whPp   = (_Float16*)carve((size_t)416 * SE * 2);
  _Float16* whFl0  = (_Float16*)carve((size_t)SE * SE * 2);
  _Float16* whFl1  = (_Float16*)carve((size_t)SE * SE * 2);
  float* HB1       = (float*)carve((size_t)NRELS * SH * sizeof(float));
  float* HB2       = (float*)carve((size_t)NRELS * SH * sizeof(float));
  float* rel_in     = HB2;  // consumed before HB2's first GEMM write
  float* effect_rel = HB2;  // HB2 free after relation_encode lands in HB1
  float* F1 = agg;          // reuse after propagation
  float* F2 = Ptmp;

  const int T = 256;

  // ---- pack all weights to padded f16 (tiny)
  pack_w(stream, pe_w0, whPe0, 200, SE, 16,  0, 13, 0,  0,0,0, 0,0,0);
  pack_w(stream, pe_w1, whPe1, 200, SE, SE,  0, 200, 0, 0,0,0, 0,0,0);
  pack_w(stream, re_w0, whRe0, 300, SH, 32,  0, 27, 0,  0,0,0, 0,0,0);
  pack_w(stream, re_w1, whRe1, 300, SH, SH,  0, 300, 0, 0,0,0, 0,0,0);
  pack_w(stream, re_w2, whRe2, 300, SH, SH,  0, 300, 0, 0,0,0, 0,0,0);
  pack_w(stream, rp_w,  whRp,  200, SE, 720, 0, 300, 0, 304, 200, 300, 512, 200, 500);
  pack_w(stream, pp_w,  whPp,  200, SE, 416, 0, 200, 0, 208, 200, 200, 0,0,0);
  pack_w(stream, fl_w0, whFl0, 200, SE, SE,  0, 200, 0, 0,0,0, 0,0,0);
  pack_w(stream, fl_w1, whFl1, 200, SE, SE,  0, 200, 0, 0,0,0, 0,0,0);

  // ---- build padded inputs
  k_mean_state<<<1, T, 0, stream>>>(state, mean6);
  k_attr_full<<<(NPART + T - 1) / T, T, 0, stream>>>(attr, state, mean6, attr_full);
  k_pe_in<<<(NPART + T - 1) / T, T, 0, stream>>>(attr_full, state, node_r, pe_in);
  k_relidx<<<(NRELS + T - 1) / T, T, 0, stream>>>(node_r, node_s, rel_recv, rel_send, ridx, sidx);
  k_rel_in<<<(NRELS + T - 1) / T, T, 0, stream>>>(attr_full, state, Ra, ridx, sidx, rel_in);

  Seg dummy{pe_in, nullptr, 0, SP};  // never selected when vend chain is tight

  // ---- particle encoder: [N,13] -> 200 -> 200
  gemm(stream, Seg{pe_in, nullptr, 16, SP}, Seg{pe_in, nullptr, 16, SP}, dummy,
       whPe0, pe_b0, Ptmp, nullptr, NPART, 200, SE, 16, 1);
  gemm(stream, Seg{Ptmp, nullptr, SE, SE}, Seg{Ptmp, nullptr, SE, SE}, dummy,
       whPe1, pe_b1, part_enc, nullptr, NPART, 200, SE, SE, 1);

  // ---- relation encoder: [NREL,27] -> 300 -> 300 -> 300
  gemm(stream, Seg{rel_in, nullptr, 32, SR}, Seg{rel_in, nullptr, 32, SR}, dummy,
       whRe0, re_b0, HB1, nullptr, NRELS, 300, SH, 32, 1);
  gemm(stream, Seg{HB1, nullptr, SH, SH}, Seg{HB1, nullptr, SH, SH}, dummy,
       whRe1, re_b1, HB2, nullptr, NRELS, 300, SH, SH, 1);
  gemm(stream, Seg{HB2, nullptr, SH, SH}, Seg{HB2, nullptr, SH, SH}, dummy,
       whRe2, re_b2, HB1, nullptr, NRELS, 300, SH, SH, 1);  // rel_enc = HB1

  // ---- propagation, pstep = 2
  k_zero<<<2048, T, 0, stream>>>(part_eff, (long)NPART * SE);
  for (int ps = 0; ps < PSTEPS; ++ps) {
    gemm(stream, Seg{HB1, nullptr, 304, SH}, Seg{part_eff, ridx, 512, SE},
         Seg{part_eff, sidx, 720, SE}, whRp, rp_b, effect_rel, nullptr, NRELS,
         200, SE, 720, 1);
    k_zero<<<2048, T, 0, stream>>>(agg, (long)NPART * SE);
    long total = (long)NRELS * SE;
    k_scatter_add<<<(unsigned)((total + T - 1) / T), T, 0, stream>>>(effect_rel, rel_recv, agg, total);
    gemm(stream, Seg{part_enc, nullptr, 208, SE}, Seg{agg, nullptr, 416, SE},
         dummy, whPp, pp_b, part_eff, node_r, NPART, 200, SE, 416, 1);
  }

  // ---- rigid head
  k_rigid_head<<<1, T, 0, stream>>>(part_eff, rg_w0, rg_b0, rg_w1, rg_b1, rg_w2,
                                    rg_b2, mean6, std_p, mean_p, smallb);
  k_rigid_out<<<(BRIGID + T - 1) / T, T, 0, stream>>>(state, smallb, std_p, mean_p,
                                                      mean_v, std_v, out);

  // ---- fluid head: 200 -> 200 -> 200 -> 3
  const float* effFluid = part_eff + (size_t)BRIGID * SE;
  gemm(stream, Seg{effFluid, nullptr, SE, SE}, Seg{effFluid, nullptr, SE, SE},
       dummy, whFl0, fl_b0, F1, nullptr, NFLUID, 200, SE, SE, 1);
  gemm(stream, Seg{F1, nullptr, SE, SE}, Seg{F1, nullptr, SE, SE}, dummy, whFl1,
       fl_b1, F2, nullptr, NFLUID, 200, SE, SE, 1);
  k_fluid_out<<<(NFLUID * 3 + T - 1) / T, T, 0, stream>>>(F2, fl_w2, fl_b2, out);
}
